// ReprogrammingLayer_GQA_35622458753686
// MI455X (gfx1250) — matrix operand order repaired and checked by reading, not compiled, not run on hardware
//
#include <hip/hip_runtime.h>
#include <hip/hip_bf16.h>
#include <cmath>

// ---------------------------------------------------------------------------
// Problem constants (ReprogrammingLayer GQA)
// ---------------------------------------------------------------------------
constexpr int Bc = 8, Lc = 1024, Sreal = 1000, Spad = 1024;
constexpr int Hc = 8, Gc = 4, Ec = 128, Dmodel = 1024, Dllm = 4096;
constexpr int BL = Bc * Lc;               // 8192 rows
constexpr int HE = Hc * Ec;               // 1024
constexpr float QK_SCALE = 0.08838834764831845f; // 1/sqrt(128)

// ---------------------------------------------------------------------------
// WMMA types / helpers (CDNA5 wave32, v_wmma_f32_16x16x32_bf16)
// ---------------------------------------------------------------------------
typedef __attribute__((ext_vector_type(16))) __bf16 bf16x16;
typedef __attribute__((ext_vector_type(8)))  __bf16 bf16x8;
typedef __attribute__((ext_vector_type(8)))  float  f32x8;
typedef __attribute__((ext_vector_type(4)))  int    v4i;

union FragAB { bf16x16 v16; bf16x8 v8[2]; };
union FragC  { f32x8 v; float f[8]; };

__device__ __forceinline__ f32x8 wmma_bf16(bf16x16 a, bf16x16 b, f32x8 c) {
  // 8 args: (neg_a, A, neg_b, B, c_mod, C, reuse_a, reuse_b)
  return __builtin_amdgcn_wmma_f32_16x16x32_bf16(false, a, false, b, (short)0, c,
                                                 false, false);
}

// A-fragment: row-major [M,K] bf16. Lane: m = (lane&15),
// k in [k0+16g, k0+16g+16)  -> two b128 loads.
__device__ __forceinline__ FragAB load_a(const __bf16* base /*row ptr + k0 + 16g*/) {
  FragAB f;
  f.v8[0] = *(const bf16x8*)(base);
  f.v8[1] = *(const bf16x8*)(base + 8);
  return f;
}
// B-fragment from transposed storage Bt[N,K]: lane n = n0+(lane&15),
// k in [k0+16g, k0+16g+16) -> two b128 loads.
__device__ __forceinline__ FragAB load_b(const __bf16* base /*row ptr + k0 + 16g*/) {
  FragAB f;
  f.v8[0] = *(const bf16x8*)(base);
  f.v8[1] = *(const bf16x8*)(base + 8);
  return f;
}

// ---------------------------------------------------------------------------
// CDNA5 async global->LDS copy (ASYNCcnt path) with compile-safe fallback
// ---------------------------------------------------------------------------
#if defined(__HIP_DEVICE_COMPILE__) && defined(__gfx1250__) && \
    __has_builtin(__builtin_amdgcn_global_load_async_to_lds_b128)
#define HAS_ASYNC_LDS 1
#else
#define HAS_ASYNC_LDS 0
#endif

__device__ __forceinline__ void async_copy16(const __bf16* g, __bf16* l) {
#if HAS_ASYNC_LDS
  // Signature (from hipcc diagnostic): param0 = v4i in AS1 (global), dst in AS3.
  __builtin_amdgcn_global_load_async_to_lds_b128(
      (__attribute__((address_space(1))) v4i*)const_cast<__bf16*>(g),
      (__attribute__((address_space(3))) v4i*)l, 0, 0);
#else
  *(bf16x8*)l = *(const bf16x8*)g;   // sync fallback: still LDS-staged
#endif
}

template <int N>
__device__ __forceinline__ void wait_async() {
#if HAS_ASYNC_LDS
#if __has_builtin(__builtin_amdgcn_s_wait_asynccnt)
  __builtin_amdgcn_s_wait_asynccnt(N);
#else
  asm volatile("s_wait_asynccnt %0" ::"n"(N) : "memory");
#endif
#endif
}

// ---------------------------------------------------------------------------
// Elementwise prep kernels (f32 -> bf16, transposes, GQA weight folding)
// ---------------------------------------------------------------------------
__global__ __launch_bounds__(256) void cvt_bf16_kernel(const float* __restrict__ src,
                                                       __bf16* __restrict__ dst,
                                                       size_t n) {
  size_t i = (size_t)blockIdx.x * 256 + threadIdx.x;
  if (i < n) dst[i] = (__bf16)src[i];
}

// src [K x N] f32 -> dst [N x K] bf16
__global__ __launch_bounds__(256) void transpose_bf16_kernel(const float* __restrict__ src,
                                                             __bf16* __restrict__ dst,
                                                             int K, int N) {
  size_t i = (size_t)blockIdx.x * 256 + threadIdx.x;
  if (i >= (size_t)K * N) return;
  int n = (int)(i / K), k = (int)(i % K);
  dst[i] = (__bf16)src[(size_t)k * N + n];
}

// Wk [Dllm x (G*E)] -> WT_sum [E x Dllm] bf16 with WT_sum[e,d] = sum_g Wk[d, g*E+e]
__global__ __launch_bounds__(256) void fold_w_kernel(const float* __restrict__ W,
                                                     __bf16* __restrict__ WT) {
  size_t i = (size_t)blockIdx.x * 256 + threadIdx.x;
  if (i >= (size_t)Ec * Dllm) return;
  int e = (int)(i / Dllm), d = (int)(i % Dllm);
  const float* row = W + (size_t)d * (Gc * Ec) + e;
  WT[i] = (__bf16)(row[0] + row[Ec] + row[2 * Ec] + row[3 * Ec]);
}

__global__ __launch_bounds__(128) void fold_b_kernel(const float* __restrict__ b,
                                                     float* __restrict__ bs) {
  int e = threadIdx.x;
  bs[e] = b[e] + b[Ec + e] + b[2 * Ec + e] + b[3 * Ec + e];
}

// ---------------------------------------------------------------------------
// Generic bf16 WMMA GEMM: C = scale * (A[M,K] @ Bt[N,K]^T + bias[N])
// Block = 256 threads (8 waves) computing a 128(M) x 128(N) tile.
// The shared 128x32 B-tile is staged in LDS via async global->LDS copies,
// double-buffered across the K loop (ASYNCcnt), removing the 8x redundant
// per-wave B loads. Grid: (N/128, Mpad/128). Rows >= M_real zero-filled.
// ---------------------------------------------------------------------------
template <bool STORE_BF16, bool STORE_T>
__global__ __launch_bounds__(256)
void gemm_bf16_kernel(const __bf16* __restrict__ A, const __bf16* __restrict__ Bt,
                      const float* __restrict__ bias, void* __restrict__ C,
                      int M_real, int lda, int K, int ldc, float scale) {
  __shared__ __bf16 bbuf[2][128 * 32];   // 2 x 8 KB

  const int tid = threadIdx.x;
  const int lane = tid & 31;
  const int wave = tid >> 5;
  const int m = lane & 15, g = lane >> 4;
  const int m0 = blockIdx.y * 128 + wave * 16;
  const int n_blk = blockIdx.x * 128;

  // B-tile copy: 128 rows x 64B = 512 chunks of 16B; thread t moves chunks
  // {t, t+256}: row = chunk>>2, 16B-slot = chunk&3.
  auto copy_tile = [&](int buf, int k0) {
#pragma unroll
    for (int c = 0; c < 2; ++c) {
      const int chunk = tid + c * 256;
      const int row = chunk >> 2;
      const int off = (chunk & 3) * 8;          // element offset within 32
      const __bf16* gp = Bt + (size_t)(n_blk + row) * K + k0 + off;
      async_copy16(gp, &bbuf[buf][row * 32 + off]);
    }
  };

  int arow = m0 + m;
  if (arow >= M_real) arow = M_real - 1;        // clamp (padded tiles)
  const __bf16* aptr = A + (size_t)arow * lda + 16 * g;

  FragC acc[8];
#pragma unroll
  for (int nt = 0; nt < 8; ++nt)
#pragma unroll
    for (int r = 0; r < 8; ++r) acc[nt].f[r] = 0.0f;

  const int nIter = K / 32;
  copy_tile(0, 0);                              // prologue
  for (int it = 0; it < nIter; ++it) {
    const int k0 = it * 32;
    if (it + 1 < nIter) {
      copy_tile((it + 1) & 1, k0 + 32);
      wait_async<2>();                          // current tile landed, next in flight
    } else {
      wait_async<0>();
    }
    __syncthreads();

    FragAB a = load_a(aptr + k0);
    const __bf16* bb = bbuf[it & 1];
#pragma unroll
    for (int nt = 0; nt < 8; ++nt) {
      const __bf16* bp = bb + (nt * 16 + m) * 32 + 16 * g;
      FragAB b;
      b.v8[0] = *(const bf16x8*)bp;
      b.v8[1] = *(const bf16x8*)(bp + 8);
      acc[nt].v = wmma_bf16(a.v16, b.v16, acc[nt].v);
    }
    __syncthreads();                            // all waves done with this buffer
  }

#pragma unroll
  for (int nt = 0; nt < 8; ++nt) {
    const int n = n_blk + nt * 16 + m;
    const float bval = bias[n];
#pragma unroll
    for (int r = 0; r < 8; ++r) {
      const int mm = m0 + r + 8 * g;            // D layout: m = r + 8*(lane/16)
      float v = (mm < M_real) ? (acc[nt].f[r] + bval) * scale : 0.0f;
      if (STORE_T) {
        ((__bf16*)C)[(size_t)n * ldc + mm] = (__bf16)v;
      } else if (STORE_BF16) {
        ((__bf16*)C)[(size_t)mm * ldc + n] = (__bf16)v;
      } else {
        ((float*)C)[(size_t)mm * ldc + n] = v;
      }
    }
  }
}

// ---------------------------------------------------------------------------
// Flash attention per (b, h, 16-row L tile). One wave (32 threads) per block.
// scores(16 x Spad) = Qh(16x128) @ Ksum^T, online softmax, O += P @ Vsum.
// Q is pre-scaled by 1/sqrt(E). Output written to head slot h_new (GQA permute).
// ---------------------------------------------------------------------------
__global__ __launch_bounds__(32)
void attention_kernel(const __bf16* __restrict__ Qbf,     // [BL, HE] (scaled)
                      const __bf16* __restrict__ Ksum,    // [Spad, E]
                      const __bf16* __restrict__ VsumT,   // [E, Spad]
                      __bf16* __restrict__ Obf) {         // [BL, HE]
  __shared__ __bf16 pbuf[16 * 32];

  const int lane = threadIdx.x & 31;
  const int m = lane & 15, g = lane >> 4;
  const int lt = blockIdx.x;                 // L tile (0..63)
  const int hq = blockIdx.y;                 // query head
  const int b  = blockIdx.z;
  const int hn = (hq & 1) * Gc + (hq >> 1);  // permuted output head

  // Q fragments for K = 128 (4 chunks of 32)
  FragAB aq[4];
  const __bf16* qrow = Qbf + ((size_t)(b * Lc + lt * 16 + m)) * HE + hq * Ec + 16 * g;
#pragma unroll
  for (int kc = 0; kc < 4; ++kc) aq[kc] = load_a(qrow + kc * 32);

  FragC acco[8];
  float mrow[8], lrow[8];
#pragma unroll
  for (int t = 0; t < 8; ++t)
#pragma unroll
    for (int r = 0; r < 8; ++r) acco[t].f[r] = 0.0f;
#pragma unroll
  for (int r = 0; r < 8; ++r) { mrow[r] = -INFINITY; lrow[r] = 0.0f; }

  for (int sc = 0; sc < Spad / 32; ++sc) {
    const int sbase = sc * 32;

    FragC s0, s1;
#pragma unroll
    for (int r = 0; r < 8; ++r) { s0.f[r] = 0.0f; s1.f[r] = 0.0f; }
#pragma unroll
    for (int kc = 0; kc < 4; ++kc) {
      const __bf16* kp0 = Ksum + (size_t)(sbase + m) * Ec      + kc * 32 + 16 * g;
      const __bf16* kp1 = Ksum + (size_t)(sbase + 16 + m) * Ec + kc * 32 + 16 * g;
      FragAB b0 = load_b(kp0);
      FragAB b1 = load_b(kp1);
      s0.v = wmma_bf16(aq[kc].v16, b0.v16, s0.v);
      s1.v = wmma_bf16(aq[kc].v16, b1.v16, s1.v);
    }

    // mask padded source positions (s >= 1000)
    if (sbase + m >= Sreal) {
#pragma unroll
      for (int r = 0; r < 8; ++r) s0.f[r] = -INFINITY;
    }
    if (sbase + 16 + m >= Sreal) {
#pragma unroll
      for (int r = 0; r < 8; ++r) s1.f[r] = -INFINITY;
    }

    // online softmax row stats; row of f[r] is (r + 8g), columns across 16 lanes
#pragma unroll
    for (int r = 0; r < 8; ++r) {
      float cm = fmaxf(s0.f[r], s1.f[r]);
      cm = fmaxf(cm, __shfl_xor(cm, 1));
      cm = fmaxf(cm, __shfl_xor(cm, 2));
      cm = fmaxf(cm, __shfl_xor(cm, 4));
      cm = fmaxf(cm, __shfl_xor(cm, 8));
      const float mn = fmaxf(mrow[r], cm);
      const float corr = __expf(mrow[r] - mn);
      mrow[r] = mn;
      const float p0 = __expf(s0.f[r] - mn);
      const float p1 = __expf(s1.f[r] - mn);
      float ps = p0 + p1;
      ps += __shfl_xor(ps, 1);
      ps += __shfl_xor(ps, 2);
      ps += __shfl_xor(ps, 4);
      ps += __shfl_xor(ps, 8);
      lrow[r] = lrow[r] * corr + ps;
#pragma unroll
      for (int t = 0; t < 8; ++t) acco[t].f[r] *= corr;
      // stage P tile (D layout -> LDS row-major 16x32)
      pbuf[(r + 8 * g) * 32 + m]      = (__bf16)p0;
      pbuf[(r + 8 * g) * 32 + 16 + m] = (__bf16)p1;
    }
    __syncthreads();

    // reload P as bf16 A-fragment (16 x 32)
    FragAB ap;
    ap.v8[0] = *(const bf16x8*)(&pbuf[m * 32 + 16 * g]);
    ap.v8[1] = *(const bf16x8*)(&pbuf[m * 32 + 16 * g + 8]);

#pragma unroll
    for (int t = 0; t < 8; ++t) {
      const __bf16* vp = VsumT + (size_t)(t * 16 + m) * Spad + sbase + 16 * g;
      FragAB bv = load_b(vp);
      acco[t].v = wmma_bf16(ap.v16, bv.v16, acco[t].v);
    }
    __syncthreads();
  }

  // normalize and store (head-permuted slot hn)
  float inv[8];
#pragma unroll
  for (int r = 0; r < 8; ++r) inv[r] = 1.0f / lrow[r];
#pragma unroll
  for (int t = 0; t < 8; ++t) {
#pragma unroll
    for (int r = 0; r < 8; ++r) {
      const size_t row = (size_t)(b * Lc + lt * 16 + r + 8 * g);
      Obf[row * HE + hn * Ec + t * 16 + m] = (__bf16)(acco[t].f[r] * inv[r]);
    }
  }
}

// ---------------------------------------------------------------------------
// Host launcher
// ---------------------------------------------------------------------------
extern "C" void kernel_launch(void* const* d_in, const int* in_sizes, int n_in,
                              void* d_out, int out_size, void* d_ws, size_t ws_size,
                              hipStream_t stream) {
  (void)in_sizes; (void)n_in; (void)out_size; (void)ws_size;
  const float* tgt = (const float*)d_in[0];
  const float* src = (const float*)d_in[1];
  const float* val = (const float*)d_in[2];
  const float* Wq  = (const float*)d_in[3];
  const float* bq  = (const float*)d_in[4];
  const float* Wk  = (const float*)d_in[5];
  const float* bk  = (const float*)d_in[6];
  const float* Wv  = (const float*)d_in[7];
  const float* bv  = (const float*)d_in[8];
  const float* Wo  = (const float*)d_in[9];
  const float* bo  = (const float*)d_in[10];
  float* out = (float*)d_out;

  char* ws = (char*)d_ws;
  size_t off = 0;
  auto nb = [&](size_t bytes) -> void* {
    off = (off + 255) & ~(size_t)255;
    void* p = ws + off;
    off += bytes;
    return p;
  };

  __bf16* tgt_bf  = (__bf16*)nb((size_t)BL * Dmodel * 2);
  __bf16* src_bf  = (__bf16*)nb((size_t)Sreal * Dllm * 2);
  __bf16* val_bf  = (__bf16*)nb((size_t)Sreal * Dllm * 2);
  __bf16* WqT_bf  = (__bf16*)nb((size_t)Dmodel * Dmodel * 2);   // [N=1024, K=1024]
  __bf16* WoT_bf  = (__bf16*)nb((size_t)Dllm * HE * 2);         // [N=4096, K=1024]
  __bf16* WkTs_bf = (__bf16*)nb((size_t)Ec * Dllm * 2);         // [128, 4096]
  __bf16* WvTs_bf = (__bf16*)nb((size_t)Ec * Dllm * 2);
  float*  bks     = (float*)nb(Ec * 4);
  float*  bvs     = (float*)nb(Ec * 4);
  __bf16* Q_bf    = (__bf16*)nb((size_t)BL * HE * 2);
  __bf16* Ksum_bf = (__bf16*)nb((size_t)Spad * Ec * 2);         // [Spad, 128]
  __bf16* VsumT   = (__bf16*)nb((size_t)Ec * Spad * 2);         // [128, Spad]
  __bf16* O_bf    = (__bf16*)nb((size_t)BL * HE * 2);

  // ---- stage 0: precision conversion + weight folding ----
  {
    size_t n = (size_t)BL * Dmodel;
    cvt_bf16_kernel<<<dim3((unsigned)((n + 255) / 256)), 256, 0, stream>>>(tgt, tgt_bf, n);
    n = (size_t)Sreal * Dllm;
    cvt_bf16_kernel<<<dim3((unsigned)((n + 255) / 256)), 256, 0, stream>>>(src, src_bf, n);
    cvt_bf16_kernel<<<dim3((unsigned)((n + 255) / 256)), 256, 0, stream>>>(val, val_bf, n);
  }
  transpose_bf16_kernel<<<dim3((Dmodel * Dmodel + 255) / 256), 256, 0, stream>>>(
      Wq, WqT_bf, Dmodel, Dmodel);
  transpose_bf16_kernel<<<dim3((HE * Dllm + 255) / 256), 256, 0, stream>>>(
      Wo, WoT_bf, HE, Dllm);
  fold_w_kernel<<<dim3((Ec * Dllm + 255) / 256), 256, 0, stream>>>(Wk, WkTs_bf);
  fold_w_kernel<<<dim3((Ec * Dllm + 255) / 256), 256, 0, stream>>>(Wv, WvTs_bf);
  fold_b_kernel<<<1, 128, 0, stream>>>(bk, bks);
  fold_b_kernel<<<1, 128, 0, stream>>>(bv, bvs);

  // ---- stage 1: projections (WMMA GEMMs, async-LDS staged B) ----
  // Q = scale * (tgt @ Wq + bq)   [8192 x 1024]
  gemm_bf16_kernel<true, false><<<dim3(HE / 128, BL / 128), 256, 0, stream>>>(
      tgt_bf, WqT_bf, bq, Q_bf, BL, Dmodel, Dmodel, HE, QK_SCALE);
  // Ksum = src @ Wk_fold + bk_fold   [Spad x 128], zero-filled rows >= 1000
  gemm_bf16_kernel<true, false><<<dim3(1, Spad / 128), 256, 0, stream>>>(
      src_bf, WkTs_bf, bks, Ksum_bf, Sreal, Dllm, Dllm, Ec, 1.0f);
  // VsumT = (val @ Wv_fold + bv_fold)^T   [128 x Spad]
  gemm_bf16_kernel<true, true><<<dim3(1, Spad / 128), 256, 0, stream>>>(
      val_bf, WvTs_bf, bvs, VsumT, Sreal, Dllm, Dllm, Spad, 1.0f);

  // ---- stage 2: GQA flash attention ----
  attention_kernel<<<dim3(Lc / 16, Hc, Bc), 32, 0, stream>>>(Q_bf, Ksum_bf, VsumT, O_bf);

  // ---- stage 3: output projection (fp32 result) ----
  gemm_bf16_kernel<false, false><<<dim3(Dllm / 128, BL / 128), 256, 0, stream>>>(
      O_bf, WoT_bf, bo, out, BL, HE, HE, Dllm, 1.0f);
}
